// ViT_MoE_29652454212577
// MI455X (gfx1250) — matrix operand order repaired and checked by reading, not compiled, not run on hardware
//
#include <hip/hip_runtime.h>
#include <hip/hip_bf16.h>
#include <math.h>

// ---------------------------------------------------------------------------
// ViT-Base + MoE forward for MI455X (gfx1250, wave32, WMMA).
// All GEMMs use v_wmma_f32_16x16x32_f16 (f16 in, f32 accumulate), register-
// blocked 2Mx2N per wave (4 WMMAs per 32-wide K step).
// ---------------------------------------------------------------------------

typedef __attribute__((ext_vector_type(16))) _Float16 v16h;
typedef __attribute__((ext_vector_type(8)))  float    v8f;

#define DD     768
#define TT     197
#define TP     224      // padded T (multiple of 32) for att @ V K-dim
#define BB     64
#define NHH    12
#define DH     64
#define HIDD   3072
#define NTOK   (BB*TT)      // 12608
#define NPATCH (BB*196)     // 12544
#define EE     8
#define CAP    1970         // int(1.25 * 12608 / 8)
#define CAPP   1984         // CAP rounded up to multiple of 16
#define NCC    1000

__device__ __forceinline__ float gelu_f(float v) {
  return 0.5f * v * (1.0f + erff(v * 0.70710678118654752f));
}

// ---------------------------------------------------------------------------
// Generic strided-batched GEMM:  C = act(alpha * A @ W^T + bias)
//   A: [M x K] f32 row-major (lda), W: [N x K] f32 row-major (ldw)
//   batch index bh -> (bi = bh/H, hi = bh%H), offsets bi*s?b + hi*s?h
//   act: 0 = none, 1 = exact GELU
// Each wave owns a 32x32 output macro-tile (2x2 of 16x16 WMMA tiles);
// 4 waves per block cover 4 consecutive 32-wide N strips.
// K must be a multiple of 32. Edge M/N handled by clamped loads + guarded
// stores (wave-uniform control flow only, EXEC all-ones at the WMMAs).
// f32 operands are converted to f16 in-register (v_cvt_pk_f16_f32 co-executes
// with the XDL WMMA pipe).
// ---------------------------------------------------------------------------
__global__ void gemm_wmma_kernel(
    const float* __restrict__ A, const float* __restrict__ W,
    const float* __restrict__ bias, float* __restrict__ C,
    int M, int N, int K, int lda, int ldw, int ldc,
    int H,
    long long sAb, long long sAh, long long sWb, long long sWh,
    long long sCb, long long sCh, long long sBias,
    float alpha, int act)
{
  const int bh = blockIdx.y;
  const int bi = bh / H;
  const int hi = bh - bi * H;
  A += (long long)bi * sAb + (long long)hi * sAh;
  W += (long long)bi * sWb + (long long)hi * sWh;
  C += (long long)bi * sCb + (long long)hi * sCh;
  if (bias) bias += (long long)bi * sBias;

  const int ntm   = (M + 15) >> 4;
  const int ntn   = (N + 15) >> 4;
  const int ntn2  = (ntn + 1) >> 1;            // 32-wide N strips
  const int ntn2w = (ntn2 + 3) >> 2;           // strips per block (4 waves)
  const int wid   = threadIdx.x >> 5;
  const int bm    = blockIdx.x / ntn2w;        // 32-wide M strip
  const int bn    = (blockIdx.x - bm * ntn2w) * 4 + wid;
  if (bn >= ntn2) return;                      // wave-uniform

  const int lane = threadIdx.x & 31;
  const int half = lane >> 4;                  // 0: lanes 0-15, 1: lanes 16-31
  const int l16  = lane & 15;

  int row0 = bm * 32 + l16;      if (row0 > M - 1) row0 = M - 1;
  int row1 = bm * 32 + 16 + l16; if (row1 > M - 1) row1 = M - 1;
  int col0 = bn * 32 + l16;      if (col0 > N - 1) col0 = N - 1;
  int col1 = bn * 32 + 16 + l16; if (col1 > N - 1) col1 = N - 1;

  const float* __restrict__ pa0 = A + (long long)row0 * lda;
  const float* __restrict__ pa1 = A + (long long)row1 * lda;
  const float* __restrict__ pw0 = W + (long long)col0 * ldw;
  const float* __restrict__ pw1 = W + (long long)col1 * ldw;

  v8f acc00 = {0.f,0.f,0.f,0.f,0.f,0.f,0.f,0.f};
  v8f acc01 = acc00, acc10 = acc00, acc11 = acc00;

  for (int k0 = 0; k0 < K; k0 += 32) {
    if (k0 + 64 < K) {                         // pull K step two ahead toward GL2
      __builtin_prefetch(pa0 + k0 + 64, 0, 1);
      __builtin_prefetch(pa1 + k0 + 64, 0, 1);
      __builtin_prefetch(pw0 + k0 + 64, 0, 1);
      __builtin_prefetch(pw1 + k0 + 64, 0, 1);
    }
    v16h af0, af1, bf0, bf1;
    // A fragment (16x32 f16): lanes 0-15 K=[k0..k0+15],
    // lanes 16-31 K=[k0+16..k0+31].
    const int ka0 = k0 + half * 16;
    const int ka1 = k0 + half * 16 + 8;
#pragma unroll
    for (int i = 0; i < 8; ++i) {
      af0[i]     = (_Float16)pa0[ka0 + i];
      af0[i + 8] = (_Float16)pa0[ka1 + i];
      af1[i]     = (_Float16)pa1[ka0 + i];
      af1[i + 8] = (_Float16)pa1[ka1 + i];
    }
    // B fragment (32x16): lane holds one column, 16 contiguous K.
    const int kb = k0 + half * 16;
#pragma unroll
    for (int i = 0; i < 16; ++i) {
      bf0[i] = (_Float16)pw0[kb + i];
      bf1[i] = (_Float16)pw1[kb + i];
    }

    acc00 = __builtin_amdgcn_wmma_f32_16x16x32_f16(false, af0, false, bf0,
                                                   (short)0, acc00, false, false);
    acc01 = __builtin_amdgcn_wmma_f32_16x16x32_f16(false, af0, false, bf1,
                                                   (short)0, acc01, false, false);
    acc10 = __builtin_amdgcn_wmma_f32_16x16x32_f16(false, af1, false, bf0,
                                                   (short)0, acc10, false, false);
    acc11 = __builtin_amdgcn_wmma_f32_16x16x32_f16(false, af1, false, bf1,
                                                   (short)0, acc11, false, false);
  }

  // C layout: VGPR r, lanes 0-15 -> (M = r, N = lane); lanes 16-31 -> (M = 8+r).
  const int n0 = bn * 32 + l16;
  const int n1 = bn * 32 + 16 + l16;
  float bias0 = (bias && n0 < N) ? bias[n0] : 0.f;
  float bias1 = (bias && n1 < N) ? bias[n1] : 0.f;
#pragma unroll
  for (int r = 0; r < 8; ++r) {
    int m0 = bm * 32 + r + half * 8;
    int m1 = m0 + 16;
    if (m0 < M) {
      if (n0 < N) {
        float v = acc00[r] * alpha + bias0;
        if (act == 1) v = gelu_f(v);
        C[(long long)m0 * ldc + n0] = v;
      }
      if (n1 < N) {
        float v = acc01[r] * alpha + bias1;
        if (act == 1) v = gelu_f(v);
        C[(long long)m0 * ldc + n1] = v;
      }
    }
    if (m1 < M) {
      if (n0 < N) {
        float v = acc10[r] * alpha + bias0;
        if (act == 1) v = gelu_f(v);
        C[(long long)m1 * ldc + n0] = v;
      }
      if (n1 < N) {
        float v = acc11[r] * alpha + bias1;
        if (act == 1) v = gelu_f(v);
        C[(long long)m1 * ldc + n1] = v;
      }
    }
  }
}

// ---------------------------------------------------------------------------
// Elementwise / reshape kernels
// ---------------------------------------------------------------------------
__global__ void im2col_kernel(const float* __restrict__ x, float* __restrict__ Xp)
{
  long long n = (long long)NPATCH * DD;
  for (long long i = blockIdx.x * 256ll + threadIdx.x; i < n;
       i += (long long)gridDim.x * 256) {
    int f = (int)(i % DD);
    long long tk = i / DD;
    int b = (int)(tk / 196), sidx = (int)(tk % 196);
    int py = sidx / 14, px = sidx % 14;
    int c = f >> 8, r = f & 255;
    int iy = r >> 4, ix = r & 15;
    Xp[i] = x[(((long long)b * 3 + c) * 224 + (py * 16 + iy)) * 224 + (px * 16 + ix)];
  }
}

__global__ void assemble_kernel(const float* __restrict__ patch_out,
                                const float* __restrict__ cls,
                                const float* __restrict__ pos,
                                float* __restrict__ h)
{
  long long n = (long long)BB * TT * DD;
  for (long long i = blockIdx.x * 256ll + threadIdx.x; i < n;
       i += (long long)gridDim.x * 256) {
    int d = (int)(i % DD);
    long long r = i / DD;
    int t = (int)(r % TT);
    int b = (int)(r / TT);
    float v = (t == 0) ? cls[d]
                       : patch_out[((long long)b * 196 + (t - 1)) * DD + d];
    h[i] = v + pos[(long long)t * DD + d];
  }
}

__global__ void add_kernel(float* __restrict__ x, const float* __restrict__ z,
                           long long n)
{
  for (long long i = blockIdx.x * 256ll + threadIdx.x; i < n;
       i += (long long)gridDim.x * 256)
    x[i] += z[i];
}

__global__ void layernorm_kernel(const float* __restrict__ X,
                                 const float* __restrict__ g,
                                 const float* __restrict__ b,
                                 float* __restrict__ Y)
{
  __shared__ float red[256];
  long long row = blockIdx.x;
  const float* x = X + row * DD;
  float* y = Y + row * DD;
  int tid = threadIdx.x;

  float s = 0.f;
  for (int i = tid; i < DD; i += 256) s += x[i];
  red[tid] = s; __syncthreads();
  for (int o = 128; o > 0; o >>= 1) { if (tid < o) red[tid] += red[tid + o]; __syncthreads(); }
  float mean = red[0] / (float)DD;
  __syncthreads();

  float v = 0.f;
  for (int i = tid; i < DD; i += 256) { float d = x[i] - mean; v += d * d; }
  red[tid] = v; __syncthreads();
  for (int o = 128; o > 0; o >>= 1) { if (tid < o) red[tid] += red[tid + o]; __syncthreads(); }
  float rstd = rsqrtf(red[0] / (float)DD + 1e-5f);

  for (int i = tid; i < DD; i += 256)
    y[i] = (x[i] - mean) * rstd * g[i] + b[i];
}

// Build V^T padded: vT[bh][d][t] = qkv[(b*197+t)*2304 + 1536 + h*64 + d], 0 for t>=197
__global__ void vtranspose_kernel(const float* __restrict__ qkv,
                                  float* __restrict__ vT)
{
  long long n = (long long)BB * NHH * DH * TP;
  for (long long i = blockIdx.x * 256ll + threadIdx.x; i < n;
       i += (long long)gridDim.x * 256) {
    int t = (int)(i % TP);
    long long r = i / TP;
    int d = (int)(r % DH);
    int bh = (int)(r / DH);
    int b = bh / NHH, hh = bh % NHH;
    float v = 0.f;
    if (t < TT)
      v = qkv[((long long)(b * TT + t)) * (3 * DD) + 2 * DD + hh * DH + d];
    vT[i] = v;
  }
}

// Row softmax over 197 valid elements of rows of length 224; pad -> 0.
__global__ void softmax_rows_kernel(float* __restrict__ S, long long rows)
{
  int wid = threadIdx.x >> 5, lane = threadIdx.x & 31;
  long long row = (long long)blockIdx.x * 8 + wid;
  if (row >= rows) return;
  float* p = S + row * TP;

  float mx = -1e30f;
  for (int i = lane; i < TT; i += 32) mx = fmaxf(mx, p[i]);
  for (int m = 16; m > 0; m >>= 1) mx = fmaxf(mx, __shfl_xor(mx, m, 32));

  float s = 0.f;
  for (int i = lane; i < TT; i += 32) { float e = __expf(p[i] - mx); p[i] = e; s += e; }
  for (int m = 16; m > 0; m >>= 1) s += __shfl_xor(s, m, 32);
  float inv = 1.f / s;

  for (int i = lane; i < TP; i += 32) p[i] = (i < TT) ? p[i] * inv : 0.f;
}

// ---------------------------------------------------------------------------
// MoE kernels
// ---------------------------------------------------------------------------
__global__ void router_kernel(const float* __restrict__ hln,
                              const float* __restrict__ rw,
                              const float* __restrict__ rb,
                              int* __restrict__ idx0, int* __restrict__ idx1,
                              float* __restrict__ zsum)
{
  int wid = threadIdx.x >> 5, lane = threadIdx.x & 31;
  int t = blockIdx.x * 8 + wid;
  if (t >= NTOK) return;
  const float* xr = hln + (long long)t * DD;

  float l[EE];
  for (int e = 0; e < EE; ++e) {
    const float* w = rw + e * DD;
    float s = 0.f;
    for (int i = lane; i < DD; i += 32) s += xr[i] * w[i];
    for (int m = 16; m > 0; m >>= 1) s += __shfl_xor(s, m, 32);
    l[e] = s + rb[e];
  }
  if (lane == 0) {
    float sq = 0.f;
    for (int e = 0; e < EE; ++e) sq += l[e] * l[e];
    atomicAdd(zsum, sq);
    int i1 = 0; float m1 = l[0];
    for (int e = 1; e < EE; ++e) if (l[e] > m1) { m1 = l[e]; i1 = e; }
    int i2 = -1; float m2 = -1e30f;
    for (int e = 0; e < EE; ++e) if (e != i1 && l[e] > m2) { m2 = l[e]; i2 = e; }
    idx0[t] = i1; idx1[t] = i2;
  }
}

// Per-expert capacity dispatch: exact cumsum order over token index.
__global__ void moe_dispatch_kernel(const int* __restrict__ i0,
                                    const int* __restrict__ i1,
                                    int* __restrict__ s2t,
                                    float* __restrict__ usage)
{
  __shared__ int sc[256];
  __shared__ int soff;
  const int e = blockIdx.x;
  const int tid = threadIdx.x;
  if (tid == 0) soff = 0;
  __syncthreads();
  for (int base = 0; base < NTOK; base += 256) {
    int t = base + tid;
    int m = (t < NTOK && (i0[t] == e || i1[t] == e)) ? 1 : 0;
    sc[tid] = m;
    __syncthreads();
    for (int o = 1; o < 256; o <<= 1) {          // inclusive scan
      int v = (tid >= o) ? sc[tid - o] : 0;
      __syncthreads();
      sc[tid] += v;
      __syncthreads();
    }
    int pos = soff + sc[tid] - m;                // exclusive position
    int total = sc[255];
    if (m && pos < CAP) s2t[e * CAPP + pos] = t;
    __syncthreads();
    if (tid == 0) soff += total;
    __syncthreads();
  }
  int kept = (soff < CAP) ? soff : CAP;
  if (tid == 0) usage[e] = (float)kept;
  for (int s = tid; s < CAPP; s += 256)
    if (s >= kept) s2t[e * CAPP + s] = NTOK;     // sentinel -> zero row
}

__global__ void moe_gather_kernel(const float* __restrict__ hln,
                                  const int* __restrict__ s2t,
                                  float* __restrict__ tokbuf)
{
  long long n = (long long)EE * CAPP * DD;
  for (long long i = blockIdx.x * 256ll + threadIdx.x; i < n;
       i += (long long)gridDim.x * 256) {
    int d = (int)(i % DD);
    long long r = i / DD;
    int slot = (int)(r % CAPP);
    int e = (int)(r / CAPP);
    int t = s2t[e * CAPP + slot];
    tokbuf[i] = (t < NTOK) ? hln[(long long)t * DD + d] : 0.f;
  }
}

__global__ void moe_scatter_kernel(const float* __restrict__ y,
                                   const int* __restrict__ s2t,
                                   float* __restrict__ h)
{
  long long n = (long long)EE * CAPP * DD;
  for (long long i = blockIdx.x * 256ll + threadIdx.x; i < n;
       i += (long long)gridDim.x * 256) {
    int d = (int)(i % DD);
    long long r = i / DD;
    int slot = (int)(r % CAPP);
    int e = (int)(r / CAPP);
    int t = s2t[e * CAPP + slot];
    if (t < NTOK) atomicAdd(&h[(long long)t * DD + d], y[i]);
  }
}

__global__ void moe_aux_kernel(const float* __restrict__ usage,
                               float* __restrict__ zsum,
                               float* __restrict__ auxAcc)
{
  float s = 1e-6f;
  for (int e = 0; e < EE; ++e) s += usage[e];
  float bsum = 0.f;
  for (int e = 0; e < EE; ++e) { float ld = usage[e] / s; bsum += ld * ld; }
  float balance = (float)EE * bsum;
  float z = 0.001f * (zsum[0] / (float)((long long)NTOK * EE));
  auxAcc[0] += balance + z;
  zsum[0] = 0.f;
}

__global__ void init_scalars_kernel(float* __restrict__ zsum,
                                    float* __restrict__ auxAcc)
{
  zsum[0] = 0.f;
  auxAcc[0] = 0.f;
}

__global__ void write_aux_kernel(const float* __restrict__ auxAcc,
                                 float* __restrict__ out)
{
  out[BB * NCC] = auxAcc[0];
}

// ---------------------------------------------------------------------------
// Host launcher
// ---------------------------------------------------------------------------
static inline dim3 gs_grid(long long n) {
  long long b = (n + 255) / 256;
  if (b > 1048576) b = 1048576;
  return dim3((unsigned)b, 1, 1);
}

extern "C" void kernel_launch(void* const* d_in, const int* in_sizes, int n_in,
                              void* d_out, int out_size, void* d_ws, size_t ws_size,
                              hipStream_t stream)
{
  (void)in_sizes; (void)n_in; (void)out_size; (void)ws_size;

  // ---- input tensors (setup_inputs dict order, nested dicts flattened) ----
  const float* x_in    = (const float*)d_in[0];   // [64,3,224,224]
  const float* patch_w = (const float*)d_in[1];   // [768, 768]
  const float* patch_b = (const float*)d_in[2];   // [768]
  const float* cls_p   = (const float*)d_in[3];   // [768]
  const float* pos_p   = (const float*)d_in[4];   // [197,768]

  const float* blk[12][14];
  int bidx = 5;
  for (int i = 0; i < 12; ++i) {
    int np = (i >= 6) ? 14 : 12;
    for (int j = 0; j < np; ++j) blk[i][j] = (const float*)d_in[bidx + j];
    for (int j = np; j < 14; ++j) blk[i][j] = nullptr;
    bidx += np;
  }
  // dense block j: 0 ln1_g 1 ln1_b 2 in_w 3 in_b 4 out_w 5 out_b 6 ln2_g 7 ln2_b
  //                8 w1 9 b1 10 w2 11 b2
  // moe block  j: 0..7 same, 8 rw 9 rb 10 w1 11 b1 12 w2 13 b2
  const float* norm_g = (const float*)d_in[bidx + 0];
  const float* norm_b = (const float*)d_in[bidx + 1];
  const float* head_w = (const float*)d_in[bidx + 2];
  const float* head_b = (const float*)d_in[bidx + 3];
  float* out_logits = (float*)d_out;              // [64,1000] + aux at [64000]

  // ---- workspace layout (f32 elements, regions aliased across phases) ----
  float* wsf = (float*)d_ws;
  size_t off = 0;
  auto alloc = [&](size_t elems) {
    float* p = wsf + off;
    off += (elems + 63) & ~(size_t)63;
    return p;
  };
  float* h    = alloc((size_t)NTOK * DD);             // residual stream
  float* hln  = alloc((size_t)NTOK * DD);             // LN output
  float* qkvR = alloc((size_t)NTOK * 3 * DD);         // qkv  | tokbuf+moe_y
  float* big1 = alloc((size_t)NTOK * HIDD);           // patch_out | scores | mlp_h
  float* big2 = alloc((size_t)EE * CAPP * HIDD);      // Xp | obuf+proj | moe_h1
  float* vTb  = alloc((size_t)BB * NHH * DH * TP);    // vT | mlp_out
  float* usage = alloc(8);
  float* zsum  = alloc(1);
  float* auxA  = alloc(1);
  int* idx0 = (int*)alloc(NTOK);
  int* idx1 = (int*)alloc(NTOK);
  int* s2t  = (int*)alloc((size_t)EE * CAPP);

  float* Xp        = big2;
  float* patch_out = big1;
  float* scores    = big1;                      // [B*H, 197, 224]
  float* mlp_h     = big1;                      // [NTOK, 3072]
  float* obuf      = big2;                      // [B, 197, 768]
  float* proj      = big2 + (size_t)NTOK * DD;
  float* moe_h1    = big2;                      // [E, CAPP, 3072]
  float* vT        = vTb;                       // [B*H, 64, 224]
  float* mlp_out   = vTb;
  float* tokbuf    = qkvR;                      // [E, CAPP, 768]
  float* moe_y     = qkvR + (size_t)EE * CAPP * DD;

  auto gemm = [&](const float* A, const float* W, const float* bias, float* C,
                  int M, int N, int K, int lda, int ldw, int ldc,
                  int batches, int H,
                  long long sAb, long long sAh, long long sWb, long long sWh,
                  long long sCb, long long sCh, long long sBias,
                  float alpha, int act) {
    int ntm = (M + 15) / 16, ntn = (N + 15) / 16;
    int ntm2 = (ntm + 1) / 2, ntn2 = (ntn + 1) / 2, ntn2w = (ntn2 + 3) / 4;
    dim3 grid((unsigned)(ntm2 * ntn2w), (unsigned)batches);
    gemm_wmma_kernel<<<grid, dim3(128, 1, 1), 0, stream>>>(
        A, W, bias, C, M, N, K, lda, ldw, ldc, H,
        sAb, sAh, sWb, sWh, sCb, sCh, sBias, alpha, act);
  };

  // ---- 0. scalars ----
  init_scalars_kernel<<<1, 1, 0, stream>>>(zsum, auxA);

  // ---- 1. patch embedding (im2col + GEMM) ----
  im2col_kernel<<<gs_grid((long long)NPATCH * DD), 256, 0, stream>>>(x_in, Xp);
  gemm(Xp, patch_w, patch_b, patch_out,
       NPATCH, DD, DD, DD, DD, DD,
       1, 1, 0, 0, 0, 0, 0, 0, 0, 1.0f, 0);
  assemble_kernel<<<gs_grid((long long)NTOK * DD), 256, 0, stream>>>(
      patch_out, cls_p, pos_p, h);

  // ---- 2. transformer blocks ----
  for (int i = 0; i < 12; ++i) {
    const float* const* p = blk[i];

    // LN1 -> hln
    layernorm_kernel<<<NTOK, 256, 0, stream>>>(h, p[0], p[1], hln);

    // qkv = hln @ in_w^T + in_b                [NTOK, 2304]
    gemm(hln, p[2], p[3], qkvR,
         NTOK, 3 * DD, DD, DD, DD, 3 * DD,
         1, 1, 0, 0, 0, 0, 0, 0, 0, 1.0f, 0);

    // V^T (padded to 224 tokens)
    vtranspose_kernel<<<gs_grid((long long)BB * NHH * DH * TP), 256, 0, stream>>>(
        qkvR, vT);

    // scores = (q @ k^T) / 8                   per (b,h): [197,197] in ld 224
    gemm(qkvR /*q*/, qkvR + DD /*k*/, nullptr, scores,
         TT, TT, DH, 3 * DD, 3 * DD, TP,
         BB * NHH, NHH,
         (long long)TT * 3 * DD, DH,            // A strides (b, h)
         (long long)TT * 3 * DD, DH,            // W strides (b, h)
         (long long)NHH * TT * TP, (long long)TT * TP,  // C strides
         0, 0.125f, 0);

    // softmax rows (+ zero the K padding)
    {
      long long rows = (long long)BB * NHH * TT;
      softmax_rows_kernel<<<(unsigned)((rows + 7) / 8), 256, 0, stream>>>(scores, rows);
    }

    // attn_out = scores @ V   -> obuf [B, 197, 768] (heads interleaved via C strides)
    gemm(scores, vT, nullptr, obuf,
         TT, DH, TP, TP, TP, DD,
         BB * NHH, NHH,
         (long long)NHH * TT * TP, (long long)TT * TP,   // A strides
         (long long)NHH * DH * TP, (long long)DH * TP,   // W strides
         (long long)TT * DD, DH,                         // C strides (b, h)
         0, 1.0f, 0);

    // proj = obuf @ out_w^T + out_b ; h += proj
    gemm(obuf, p[4], p[5], proj,
         NTOK, DD, DD, DD, DD, DD,
         1, 1, 0, 0, 0, 0, 0, 0, 0, 1.0f, 0);
    add_kernel<<<gs_grid((long long)NTOK * DD), 256, 0, stream>>>(
        h, proj, (long long)NTOK * DD);

    // LN2 -> hln
    layernorm_kernel<<<NTOK, 256, 0, stream>>>(h, p[6], p[7], hln);

    if (i < 6) {
      // dense FFN
      gemm(hln, p[8], p[9], mlp_h,
           NTOK, HIDD, DD, DD, DD, HIDD,
           1, 1, 0, 0, 0, 0, 0, 0, 0, 1.0f, 1 /*gelu*/);
      gemm(mlp_h, p[10], p[11], mlp_out,
           NTOK, DD, HIDD, HIDD, HIDD, DD,
           1, 1, 0, 0, 0, 0, 0, 0, 0, 1.0f, 0);
      add_kernel<<<gs_grid((long long)NTOK * DD), 256, 0, stream>>>(
          h, mlp_out, (long long)NTOK * DD);
    } else {
      // MoE: route -> dispatch -> gather -> expert FFN (batched) -> scatter
      router_kernel<<<(NTOK + 7) / 8, 256, 0, stream>>>(
          hln, p[8], p[9], idx0, idx1, zsum);
      moe_dispatch_kernel<<<EE, 256, 0, stream>>>(idx0, idx1, s2t, usage);
      moe_gather_kernel<<<gs_grid((long long)EE * CAPP * DD), 256, 0, stream>>>(
          hln, s2t, tokbuf);

      // h1 = gelu(tok @ w1[e]^T + b1[e])   [E, CAPP, 3072]
      gemm(tokbuf, p[10], p[11], moe_h1,
           CAPP, HIDD, DD, DD, DD, HIDD,
           EE, 1,
           (long long)CAPP * DD, 0,
           (long long)HIDD * DD, 0,
           (long long)CAPP * HIDD, 0,
           HIDD, 1.0f, 1 /*gelu*/);
      // y = h1 @ w2[e]^T + b2[e]           [E, CAPP, 768]
      gemm(moe_h1, p[12], p[13], moe_y,
           CAPP, DD, HIDD, HIDD, HIDD, DD,
           EE, 1,
           (long long)CAPP * HIDD, 0,
           (long long)DD * HIDD, 0,
           (long long)CAPP * DD, 0,
           DD, 1.0f, 0);

      moe_scatter_kernel<<<gs_grid((long long)EE * CAPP * DD), 256, 0, stream>>>(
          moe_y, s2t, h);
      moe_aux_kernel<<<1, 1, 0, stream>>>(usage, zsum, auxA);
    }
  }

  // ---- 3. final LN + head (cls rows picked via lda = 197*768) ----
  layernorm_kernel<<<NTOK, 256, 0, stream>>>(h, norm_g, norm_b, hln);
  gemm(hln, head_w, head_b, out_logits,
       BB, NCC, DD, TT * DD, DD, NCC,
       1, 1, 0, 0, 0, 0, 0, 0, 0, 1.0f, 0);

  // ---- 4. aux scalar ----
  write_aux_kernel<<<1, 1, 0, stream>>>(auxA, out_logits);
}
